// PatchCCASparse_12661563589196
// MI455X (gfx1250) — matrix operand order repaired and checked by reading, not compiled, not run on hardware
//
#include <hip/hip_runtime.h>
#include <hip/hip_bf16.h>

// PatchCCASparse for MI455X (gfx1250, wave32, WMMA).
// One workgroup (256 threads = 8 waves) per (batch, coordinate, radius).
// Phases: prefetch footprint (global_prefetch_b8) -> bilinear gather -> LDS (f16,
//         channel-major, K-padded) -> standardize columns (4-thread/column) ->
//         3x (32x32) covariances via v_wmma_f32_16x16x32_f16 ->
//         Cholesky + triangular inverse + |diag| of Lxx^-1 * Cxy * Lyy^-1 (f32 VALU).

#define Bc 4
#define Cc 32
#define Hc 512
#define Wc 512
#define Nc 1024
#define KS 256   // max padded sample count (stride of channel-major patch storage)

typedef __attribute__((ext_vector_type(16))) _Float16 v16h;
typedef __attribute__((ext_vector_type(8)))  float    v8f;

__global__ __launch_bounds__(256)
void patch_cca_kernel(const float* __restrict__ xg,
                      const float* __restrict__ yg,
                      const float* __restrict__ coords,
                      float* __restrict__ out) {
  __shared__ __align__(16) _Float16 Xh[Cc * KS];   // 16 KB
  __shared__ __align__(16) _Float16 Yh[Cc * KS];   // 16 KB
  __shared__ float cov[3 * 32 * 32];               // 12 KB: [0]=xx [1]=yy [2]=xy
  __shared__ float red[32];
  float* Linv = (float*)Xh;                        // reuse patch LDS after cov phase (2*1024 f32)

  const int radii[3] = {3, 5, 7};
  const int bn   = blockIdx.x;          // b*Nc + n
  const int b    = bn >> 10;
  const int ridx = blockIdx.y;
  const int r    = radii[ridx];
  const int k    = 2 * r + 1;
  const int kk   = k * k;
  const int kkPad = (kk + 31) & ~31;    // K granularity 32 for f16 WMMA
  const int tid  = threadIdx.x;

  const float c0 = coords[(bn << 1) + 0];
  const float c1 = coords[(bn << 1) + 1];
  const float dy = 1.0f / (float)(Hc - 1);
  const float dx = 1.0f / (float)(Wc - 1);

  // ---- phase 0: prefetch the whole gather footprint into cache ----
  // footprint: rows floor((c1 - r*dx)*(H-1)) .. +k, cols floor((c0 - r*dy)*(W-1)) .. +k
  {
    int xlo  = (int)floorf((c0 - (float)r * dy) * (float)(Wc - 1));
    int ylo  = (int)floorf((c1 - (float)r * dx) * (float)(Hc - 1));
    int nrow = k + 1;
    int x0c  = min(max(xlo, 0), Wc - 1);
    int x1c  = min(max(xlo + k, 0), Wc - 1);
    for (int idx = tid; idx < 2 * Cc * nrow; idx += 256) {
      int img = idx / (Cc * nrow);
      int rem = idx - img * (Cc * nrow);
      int c   = rem / nrow;
      int row = rem - c * nrow;
      int yy  = min(max(ylo + row, 0), Hc - 1);
      const float* base = (img ? yg : xg) + (size_t)(b * Cc + c) * (Hc * Wc) + yy * Wc;
      __builtin_prefetch(base + x0c, 0, 3);   // global_prefetch_b8 (row spans <= 2 lines)
      __builtin_prefetch(base + x1c, 0, 3);
    }
  }

  // ---- zero K-padding rows (they contribute 0 to X^T X) ----
  const int pad = kkPad - kk;
  for (int idx = tid; idx < pad * Cc; idx += 256) {
    int c = idx / pad;
    int s = kk + idx % pad;
    Xh[c * KS + s] = (_Float16)0.0f;
    Yh[c * KS + s] = (_Float16)0.0f;
  }

  // ---- phase 1: bilinear gather of patches into LDS ----
  // reference grid: sampler-x = c0 + lin[a]*dy, sampler-y = c1 + lin[bb]*dx
  // (sample ordering is irrelevant: mean/std/cov are permutation invariant,
  //  so we make the thread-fastest index drive sampler-x for coalesced rows)
  const int total = 2 * Cc * kk;
  for (int idx = tid; idx < total; idx += 256) {
    int img  = idx / (Cc * kk);
    int remc = idx - img * (Cc * kk);
    int c    = remc / kk;
    int s    = remc - c * kk;
    int a    = s % k;                  // fast -> sampler-x -> image column (contiguous)
    int bb   = s / k;
    float fx = (c0 + (float)(a  - r) * dy) * (float)(Wc - 1);
    float fy = (c1 + (float)(bb - r) * dx) * (float)(Hc - 1);
    float x0f = floorf(fx), y0f = floorf(fy);
    float wx1 = fx - x0f, wx0 = 1.0f - wx1;
    float wy1 = fy - y0f, wy0 = 1.0f - wy1;
    int x0 = (int)x0f, y0 = (int)y0f;
    const float* base = (img ? yg : xg) + (size_t)(b * Cc + c) * (Hc * Wc);
    auto g = [&](int yy, int xx) -> float {
      return ((unsigned)yy < (unsigned)Hc && (unsigned)xx < (unsigned)Wc)
                 ? base[yy * Wc + xx] : 0.0f;
    };
    float val = g(y0,     x0    ) * (wy0 * wx0)
              + g(y0,     x0 + 1) * (wy0 * wx1)
              + g(y0 + 1, x0    ) * (wy1 * wx0)
              + g(y0 + 1, x0 + 1) * (wy1 * wx1);
    (img ? Yh : Xh)[c * KS + s] = (_Float16)val;
  }
  __syncthreads();

  // ---- phase 2: standardize each channel over kk samples (ddof=1) ----
  // 4 threads per channel-column; partial reductions staged in cov[] (dead until
  // phase 3 fully overwrites it).
  {
    const int cid = tid >> 2;          // 0..63 (column id across both matrices)
    const int sub = tid & 3;
    _Float16* col = (cid < 32 ? Xh : Yh) + (cid & 31) * KS;
    float ps = 0.0f;
    for (int s = sub; s < kk; s += 4) ps += (float)col[s];
    cov[tid] = ps;
    __syncthreads();
    if (sub == 0)
      cov[256 + cid] = (cov[tid] + cov[tid + 1] + cov[tid + 2] + cov[tid + 3])
                       / (float)kk;
    __syncthreads();
    float mean = cov[256 + cid];
    float pv = 0.0f;
    for (int s = sub; s < kk; s += 4) { float d = (float)col[s] - mean; pv += d * d; }
    __syncthreads();                   // cov[tid] partial-sum slots reused
    cov[tid] = pv;
    __syncthreads();
    if (sub == 0) {
      float var = (cov[tid] + cov[tid + 1] + cov[tid + 2] + cov[tid + 3])
                  / (float)(kk - 1);
      cov[384 + cid] = rsqrtf(var);
    }
    __syncthreads();
    float istd = cov[384 + cid];
    for (int s = sub; s < kk; s += 4)
      col[s] = (_Float16)(((float)col[s] - mean) * istd);
  }
  __syncthreads();

  // ---- phase 3: 32x32 covariances via WMMA f16 -> f32 ----
  // 12 jobs = {xx,yy,xy} x 4 (16x16) tiles, one wave per job.
  {
    const int wave = tid >> 5, lane = tid & 31;
    const int m = lane & 15, hi = lane >> 4;
    for (int job = wave; job < 12; job += 8) {
      int p = job >> 2, t = job & 3, ti = t >> 1, tj = t & 1;
      const _Float16* As = (p == 1) ? Yh : Xh;   // xx,xy use X; yy uses Y
      const _Float16* Bs = (p == 0) ? Xh : Yh;   // xx uses X; yy,xy use Y
      const _Float16* Acol = As + (ti * 16 + m) * KS;
      const _Float16* Bcol = Bs + (tj * 16 + m) * KS;
      v8f acc = {};
      for (int kb = 0; kb < kkPad; kb += 32) {
        v16h av, bv;
#pragma unroll
        for (int v = 0; v < 8; ++v) {
          // 16-bit A 16x32 layout: VGPR v<4: K=2v(+8 for hi lanes); v>=4: K=16+2(v-4)(+8)
          int kA = kb + ((v < 4) ? 2 * v : 16 + 2 * (v - 4)) + hi * 8;
          // 16-bit B 32x16 layout (matches A): VGPR v<4: K=2v(+8 for hi lanes); v>=4: K=16+2(v-4)(+8)
          int kB = kb + ((v < 4) ? 2 * v : 16 + 2 * (v - 4)) + hi * 8;
          av[2 * v]     = Acol[kA];
          av[2 * v + 1] = Acol[kA + 1];
          bv[2 * v]     = Bcol[kB];
          bv[2 * v + 1] = Bcol[kB + 1];
        }
        acc = __builtin_amdgcn_wmma_f32_16x16x32_f16(
            /*neg_a=*/false, av, /*neg_b=*/false, bv,
            /*c_mod=*/(short)0, acc, /*reuse_a=*/false, /*reuse_b=*/false);
      }
      float scale = 1.0f / (float)(kk - 1);
#pragma unroll
      for (int v = 0; v < 8; ++v)   // C layout: VGPR v -> row v (+8 for hi lanes), col = m
        cov[p * 1024 + (ti * 16 + hi * 8 + v) * 32 + tj * 16 + m] = acc[v] * scale;
    }
  }
  __syncthreads();

  // ---- phase 4: + eps*I on cov_xx / cov_yy ----
  if (tid < 64) cov[(tid >> 5) * 1024 + (tid & 31) * 33] += 1e-6f;
  __syncthreads();

  // ---- phase 5: in-place Cholesky of both 32x32 matrices (block halves) ----
  {
    const int mat = tid >> 7;       // threads 0-127 -> xx, 128-255 -> yy
    const int lt  = tid & 127;
    float* M = cov + mat * 1024;
    for (int j = 0; j < 32; ++j) {
      if (lt == 0) M[j * 33] = sqrtf(M[j * 33]);
      __syncthreads();
      float dj = M[j * 33];
      for (int i = j + 1 + lt; i < 32; i += 128) M[i * 32 + j] /= dj;
      __syncthreads();
      for (int kc = j + 1; kc < 32; ++kc) {
        float Lkj = M[kc * 32 + j];
        for (int i = kc + lt; i < 32; i += 128)
          M[i * 32 + kc] -= M[i * 32 + j] * Lkj;
      }
      __syncthreads();
    }
  }

  // ---- phase 6: invert lower-triangular factors (one column per thread) ----
  if (tid < 64) {
    int mm = tid >> 5, j = tid & 31;
    float* L  = cov + mm * 1024;
    float* Li = Linv + mm * 1024;
    for (int i = 0; i < 32; ++i) {
      float v;
      if (i < j)       v = 0.0f;
      else if (i == j) v = 1.0f / L[i * 33];
      else {
        float s = 0.0f;
        for (int p = j; p < i; ++p) s += L[i * 32 + p] * Li[p * 32 + j];
        v = -s / L[i * 33];
      }
      Li[i * 32 + j] = v;
    }
  }
  __syncthreads();

  // ---- phase 7: diag(Lxx^-1 * Cxy * Lyy^-1), mean |.| ----
  if (tid < 32) {
    int d = tid;
    const float* A   = Linv;           // lower: A[d][i] != 0 for i <= d
    const float* Bv  = Linv + 1024;    // lower: B[j][d] != 0 for j >= d
    const float* Cxy = cov + 2048;
    float acc = 0.0f;
    for (int j = d; j < 32; ++j) {
      float t = 0.0f;
      for (int i = 0; i <= d; ++i) t += A[d * 32 + i] * Cxy[i * 32 + j];
      acc += t * Bv[j * 32 + d];
    }
    red[d] = fabsf(acc);
  }
  __syncthreads();
  if (tid == 0) {
    float s = 0.0f;
    for (int i = 0; i < 32; ++i) s += red[i];
    out[bn * 3 + ridx] = s * (1.0f / 32.0f);
  }
}

extern "C" void kernel_launch(void* const* d_in, const int* in_sizes, int n_in,
                              void* d_out, int out_size, void* d_ws, size_t ws_size,
                              hipStream_t stream) {
  const float* x      = (const float*)d_in[0];
  const float* y      = (const float*)d_in[1];
  const float* coords = (const float*)d_in[2];
  float* out = (float*)d_out;
  dim3 grid(Bc * Nc, 3);   // (b,n) x radius
  patch_cca_kernel<<<grid, 256, 0, stream>>>(x, y, coords, out);
}
